// MoHAttention_12197707121243
// MI455X (gfx1250) — matrix operand order repaired and checked by reading, not compiled, not run on hardware
//
#include <hip/hip_runtime.h>
#include <hip/hip_bf16.h>

#define DIM      1024
#define NHEADS   16
#define HEAD_DIM 64
#define BATCH    2
#define SEQ      2048
#define NTOK     (BATCH*SEQ)     // 4096 tokens
#define NBH      (BATCH*NHEADS)  // 32 (b,h) pairs

typedef __attribute__((ext_vector_type(16))) __bf16 v16bf;
typedef __attribute__((ext_vector_type(8)))  __bf16 v8bf;
typedef __attribute__((ext_vector_type(8)))  float  v8f;

union ABFrag { v16bf v; v8bf h[2]; };

__device__ inline unsigned short f2bf(float f) {
  union { float f; unsigned u; } x; x.f = f;
  unsigned r = x.u + 0x7FFFu + ((x.u >> 16) & 1u);  // round-nearest-even
  return (unsigned short)(r >> 16);
}

__device__ inline v8f wmma_bf16(v16bf a, v16bf b, v8f c) {
  return __builtin_amdgcn_wmma_f32_16x16x32_bf16(false, a, false, b, (short)0, c, false, false);
}

__device__ inline float wsum32(float v) {
#pragma unroll
  for (int off = 16; off > 0; off >>= 1) v += __shfl_xor(v, off, 32);
  return v;
}
__device__ inline float rmax16(float v) {
#pragma unroll
  for (int off = 8; off > 0; off >>= 1) v = fmaxf(v, __shfl_xor(v, off, 16));
  return v;
}
__device__ inline float rsum16(float v) {
#pragma unroll
  for (int off = 8; off > 0; off >>= 1) v += __shfl_xor(v, off, 16);
  return v;
}

// ---------------------------------------------------------------- convert
__global__ void k_f32_to_bf16(const float* __restrict__ in,
                              unsigned short* __restrict__ out, int n) {
  int i = blockIdx.x * blockDim.x + threadIdx.x;
  if (i < n) out[i] = f2bf(in[i]);
}

// ---------------------------------------------------------------- GEMM: C[M,N] = A[M,K](bf16) @ W[N,K]^T(bf16) + bias[N]
// Block: 256 thr = 8 waves; wave tile 64x32 (8 WMMAs / 12 b128 loads per K-step);
// block tile 128(M) x 128(N).
__global__ void __launch_bounds__(256)
k_gemm_bf16(const unsigned short* __restrict__ A,
            const unsigned short* __restrict__ W,
            const float* __restrict__ bias,
            float* __restrict__ C, int N, int K) {
  const int lane = threadIdx.x & 31;
  const int wave = threadIdx.x >> 5;
  const int wm = wave & 1, wn = wave >> 1;
  const int row0 = blockIdx.y * 128 + wm * 64;
  const int col0 = blockIdx.x * 128 + wn * 32;
  const int m = lane & 15, hf = lane >> 4;

  v8f acc[4][2] = {};
  const unsigned short* ap[4];
#pragma unroll
  for (int i = 0; i < 4; i++) ap[i] = A + (size_t)(row0 + i * 16 + m) * K;
  const unsigned short* w0p = W + (size_t)(col0 + m) * K + hf * 16;
  const unsigned short* w1p = W + (size_t)(col0 + 16 + m) * K + hf * 16;

  for (int kk = 0; kk < K; kk += 32) {
    v16bf b0 = *(const v16bf*)(w0p + kk);
    v16bf b1 = *(const v16bf*)(w1p + kk);
#pragma unroll
    for (int i = 0; i < 4; i++) {
      ABFrag a;
      a.h[0] = *(const v8bf*)(ap[i] + kk + hf * 16);
      a.h[1] = *(const v8bf*)(ap[i] + kk + 8 + hf * 16);
      acc[i][0] = wmma_bf16(a.v, b0, acc[i][0]);
      acc[i][1] = wmma_bf16(a.v, b1, acc[i][1]);
    }
  }
#pragma unroll
  for (int i = 0; i < 4; i++)
#pragma unroll
    for (int j = 0; j < 2; j++) {
      float bj = bias[col0 + j * 16 + m];
      float* cp = C + (size_t)(row0 + i * 16 + 8 * hf) * N + col0 + j * 16 + m;
#pragma unroll
      for (int r = 0; r < 8; r++) cp[(size_t)r * N] = acc[i][j][r] + bj;
    }
}

// ---------------------------------------------------------------- Q prep: L2-normalize + query_embedding, * softplus(temp); -> bf16 [bh,n,64]
__global__ void __launch_bounds__(256)
k_prep_q(const float* __restrict__ qf, const float* __restrict__ temp,
         const float* __restrict__ qe, unsigned int* __restrict__ qbf) {
  const int lane = threadIdx.x & 31, wave = threadIdx.x >> 5;
  const int row = blockIdx.x * 8 + wave;   // = bh*SEQ + n
  const int bh = row >> 11, n = row & 2047;
  const int b = bh >> 4, head = bh & 15;
  const float* src = qf + ((size_t)(b * SEQ + n)) * DIM + head * HEAD_DIM + lane * 2;
  float q0 = src[0], q1 = src[1];
  float inv = rsqrtf(wsum32(q0 * q0 + q1 * q1));
  float sp = log1pf(__expf(temp[head]));
  const float* qeh = qe + head * HEAD_DIM + lane * 2;
  float o0 = (q0 * inv + qeh[0]) * sp;
  float o1 = (q1 * inv + qeh[1]) * sp;
  qbf[(size_t)row * 32 + lane] = (unsigned)f2bf(o0) | ((unsigned)f2bf(o1) << 16);
}

// ---------------------------------------------------------------- K/V prep: K -> bf16 [bh,n,64]; V -> bf16 transposed [bh,64,N]
__global__ void k_prep_kv(const float* __restrict__ kf, const float* __restrict__ vf,
                          unsigned short* __restrict__ kb, unsigned short* __restrict__ vT) {
  int idx = blockIdx.x * blockDim.x + threadIdx.x;  // over NTOK*DIM
  int token = idx >> 10, c = idx & 1023;
  int head = c >> 6, d = c & 63;
  int b = token >> 11, n = token & 2047;
  int bh = b * NHEADS + head;
  kb[((size_t)bh * SEQ + n) * HEAD_DIM + d] = f2bf(kf[idx]);
  vT[((size_t)bh * HEAD_DIM + d) * SEQ + n] = f2bf(vf[idx]);
}

// ---------------------------------------------------------------- Flash attention: 1 wave per 16 q rows per (b,h); 64 keys/iter
__global__ void __launch_bounds__(32)
k_attn(const unsigned short* __restrict__ qb, const unsigned short* __restrict__ kb,
       const unsigned short* __restrict__ vT, float* __restrict__ H) {
  __shared__ unsigned short P[16 * 64];   // 2 KB: exp(S) tile, row-major 16x64
  const int lane = threadIdx.x;
  const int bh = blockIdx.x;
  const int q0row = blockIdx.y * 16;
  const int b = bh >> 4, head = bh & 15;
  const int m = lane & 15, hf = lane >> 4;

  const unsigned short* qrow = qb + ((size_t)bh * SEQ + q0row + m) * HEAD_DIM;
  ABFrag A0, A1;
  A0.h[0] = *(const v8bf*)(qrow + hf * 16);
  A0.h[1] = *(const v8bf*)(qrow + 8 + hf * 16);
  A1.h[0] = *(const v8bf*)(qrow + 32 + hf * 16);
  A1.h[1] = *(const v8bf*)(qrow + 40 + hf * 16);

  v8f O[4] = {};
  float mrun[8], lrun[8];
#pragma unroll
  for (int r = 0; r < 8; r++) { mrun[r] = -3.0e38f; lrun[r] = 0.f; }

  const unsigned short* kbase = kb + (size_t)bh * SEQ * HEAD_DIM;
  const unsigned short* vbase = vT + (size_t)bh * HEAD_DIM * SEQ;
  const float rs = 0.125f;  // 1/sqrt(64)

  for (int kt = 0; kt < SEQ; kt += 64) {
    // prefetch next 64-key K-tile into L2/L0 (global_prefetch_b8)
    if (kt + 64 < SEQ)
      __builtin_prefetch(kbase + (size_t)(kt + 64 + lane * 2) * HEAD_DIM, 0, 1);

    // S(16x64) = q(16x64) @ k^T : 8 WMMAs, 4 key sub-tiles of 16
    v8f s[4] = {};
#pragma unroll
    for (int j = 0; j < 4; j++) {
      const unsigned short* kp = kbase + (size_t)(kt + j * 16 + m) * HEAD_DIM + hf * 16;
      v16bf bk0 = *(const v16bf*)(kp);
      v16bf bk1 = *(const v16bf*)(kp + 32);
      s[j] = wmma_bf16(A0.v, bk0, s[j]);
      s[j] = wmma_bf16(A1.v, bk1, s[j]);
    }

    // online softmax; row r lives in this half's 16 lanes -> width-16 reductions
    float alpha[8];
#pragma unroll
    for (int r = 0; r < 8; r++) {
      float x0 = s[0][r] * rs, x1 = s[1][r] * rs;
      float x2 = s[2][r] * rs, x3 = s[3][r] * rs;
      float tmax = rmax16(fmaxf(fmaxf(x0, x1), fmaxf(x2, x3)));
      float mnew = fmaxf(mrun[r], tmax);
      float a = __expf(mrun[r] - mnew);
      float p0 = __expf(x0 - mnew);
      float p1 = __expf(x1 - mnew);
      float p2 = __expf(x2 - mnew);
      float p3 = __expf(x3 - mnew);
      lrun[r] = lrun[r] * a + rsum16((p0 + p1) + (p2 + p3));
      mrun[r] = mnew;
      alpha[r] = a;
      unsigned short* pr = &P[(r + 8 * hf) * 64 + m];
      pr[0]  = f2bf(p0);
      pr[16] = f2bf(p1);
      pr[32] = f2bf(p2);
      pr[48] = f2bf(p3);
    }
#pragma unroll
    for (int c = 0; c < 4; c++)
#pragma unroll
      for (int r = 0; r < 8; r++) O[c][r] *= alpha[r];

    // reload P in A-fragment layout: two 16x32 K-steps, contiguous 16B LDS runs
    ABFrag PA0, PA1;
    PA0.h[0] = *(const v8bf*)(&P[m * 64 + hf * 16]);
    PA0.h[1] = *(const v8bf*)(&P[m * 64 + 8 + hf * 16]);
    PA1.h[0] = *(const v8bf*)(&P[m * 64 + 32 + hf * 16]);
    PA1.h[1] = *(const v8bf*)(&P[m * 64 + 40 + hf * 16]);

    // O(16x64) += P(16x64) @ V(64x64): 8 WMMAs, B-frags contiguous from vT
#pragma unroll
    for (int c = 0; c < 4; c++) {
      const unsigned short* vp = vbase + (size_t)(c * 16 + m) * SEQ + kt + hf * 16;
      v16bf bv0 = *(const v16bf*)(vp);
      v16bf bv1 = *(const v16bf*)(vp + 32);
      O[c] = wmma_bf16(PA0.v, bv0, O[c]);
      O[c] = wmma_bf16(PA1.v, bv1, O[c]);
    }
  }

  float* hp = H + ((size_t)(b * SEQ) + q0row + 8 * hf) * DIM + head * HEAD_DIM + m;
#pragma unroll
  for (int c = 0; c < 4; c++)
#pragma unroll
    for (int r = 0; r < 8; r++)
      hp[(size_t)r * DIM + c * 16] = O[c][r] / lrun[r];
}

// ---------------------------------------------------------------- gating: s = 2*sw0 + 6*sw1*sum(top3); write bf16(s*h)
__global__ void __launch_bounds__(256)
k_gate(const float* __restrict__ H, const float* __restrict__ Wr, const float* __restrict__ br,
       const float* __restrict__ Ws, const float* __restrict__ bs,
       unsigned short* __restrict__ hsb) {
  const int lane = threadIdx.x & 31, wave = threadIdx.x >> 5;
  const int token = blockIdx.x * 8 + wave;
  const float* hrow = H + (size_t)token * DIM;
  float hreg[32];
#pragma unroll
  for (int t = 0; t < 32; t++) hreg[t] = hrow[lane + 32 * t];

  float lg[17];
#pragma unroll
  for (int o = 0; o < 15; o++) {
    const float* wr = Wr + (size_t)o * DIM;
    float acc = 0.f;
#pragma unroll
    for (int t = 0; t < 32; t++) acc += hreg[t] * wr[lane + 32 * t];
    lg[o] = wsum32(acc) + br[o];
  }
#pragma unroll
  for (int o = 0; o < 2; o++) {
    const float* wr = Ws + (size_t)o * DIM;
    float acc = 0.f;
#pragma unroll
    for (int t = 0; t < 32; t++) acc += hreg[t] * wr[lane + 32 * t];
    lg[15 + o] = wsum32(acc) + bs[o];
  }
  // softmax over 15 routed gates
  float gm = -3.0e38f;
#pragma unroll
  for (int o = 0; o < 15; o++) gm = fmaxf(gm, lg[o]);
  float g[15], gsum = 0.f;
#pragma unroll
  for (int o = 0; o < 15; o++) { g[o] = __expf(lg[o] - gm); gsum += g[o]; }
  float ginv = 1.f / gsum;
#pragma unroll
  for (int o = 0; o < 15; o++) g[o] *= ginv;
  // top-3 via 3-pass masked max (register-only)
  float m1 = -1.f, m2 = -1.f, m3 = -1.f;
#pragma unroll
  for (int o = 0; o < 15; o++) m1 = fmaxf(m1, g[o]);
#pragma unroll
  for (int o = 0; o < 15; o++) if (g[o] < m1) m2 = fmaxf(m2, g[o]);
#pragma unroll
  for (int o = 0; o < 15; o++) if (g[o] < m2) m3 = fmaxf(m3, g[o]);
  float sum3 = m1 + m2 + m3;
  // shared 2-way softmax
  float sm = fmaxf(lg[15], lg[16]);
  float e0 = __expf(lg[15] - sm), e1 = __expf(lg[16] - sm);
  float sw0 = e0 / (e0 + e1), sw1 = e1 / (e0 + e1);
  float s = 2.f * sw0 * (sw0 + sw1) + 6.f * sw1 * sum3;

  unsigned short* op = hsb + (size_t)token * DIM;
#pragma unroll
  for (int t = 0; t < 32; t++) op[lane + 32 * t] = f2bf(hreg[t] * s);
}

// ----------------------------------------------------------------
extern "C" void kernel_launch(void* const* d_in, const int* in_sizes, int n_in,
                              void* d_out, int out_size, void* d_ws, size_t ws_size,
                              hipStream_t stream) {
  (void)in_sizes; (void)n_in; (void)out_size; (void)ws_size;
  const float* x   = (const float*)d_in[0];
  const float* Wq  = (const float*)d_in[1];  const float* bq = (const float*)d_in[2];
  const float* Wk  = (const float*)d_in[3];  const float* bk = (const float*)d_in[4];
  const float* Wv  = (const float*)d_in[5];  const float* bv = (const float*)d_in[6];
  const float* Wp  = (const float*)d_in[7];  const float* bp = (const float*)d_in[8];
  const float* Wr  = (const float*)d_in[9];  const float* br = (const float*)d_in[10];
  const float* Ws  = (const float*)d_in[11]; const float* bs = (const float*)d_in[12];
  const float* temp = (const float*)d_in[13];
  const float* qe   = (const float*)d_in[14];
  float* out = (float*)d_out;

  char* w = (char*)d_ws;
  auto alloc = [&](size_t bytes) -> char* {
    char* p = w; w += (bytes + 255) & ~(size_t)255; return p;
  };
  const size_t NE = (size_t)NTOK * DIM;  // 4M elements
  unsigned short* xbf = (unsigned short*)alloc(NE * 2);
  unsigned short* wqb = (unsigned short*)alloc((size_t)DIM * DIM * 2);
  unsigned short* wkb = (unsigned short*)alloc((size_t)DIM * DIM * 2);
  unsigned short* wvb = (unsigned short*)alloc((size_t)DIM * DIM * 2);
  unsigned short* wpb = (unsigned short*)alloc((size_t)DIM * DIM * 2);
  float* qf = (float*)alloc(NE * 4);
  float* kf = (float*)alloc(NE * 4);
  float* vf = (float*)alloc(NE * 4);
  unsigned short* qbf = (unsigned short*)alloc(NE * 2);
  unsigned short* kbf = (unsigned short*)alloc(NE * 2);
  unsigned short* vT  = (unsigned short*)alloc(NE * 2);
  float* hbuf = (float*)alloc(NE * 4);
  unsigned short* hsb = (unsigned short*)alloc(NE * 2);

  const int nX = (int)NE, nW = DIM * DIM;
  k_f32_to_bf16<<<nX / 256, 256, 0, stream>>>(x, xbf, nX);
  k_f32_to_bf16<<<nW / 256, 256, 0, stream>>>(Wq, wqb, nW);
  k_f32_to_bf16<<<nW / 256, 256, 0, stream>>>(Wk, wkb, nW);
  k_f32_to_bf16<<<nW / 256, 256, 0, stream>>>(Wv, wvb, nW);
  k_f32_to_bf16<<<nW / 256, 256, 0, stream>>>(Wp, wpb, nW);

  dim3 gg(DIM / 128, NTOK / 128);  // (8, 32)
  k_gemm_bf16<<<gg, 256, 0, stream>>>(xbf, wqb, bq, qf, DIM, DIM);
  k_gemm_bf16<<<gg, 256, 0, stream>>>(xbf, wkb, bk, kf, DIM, DIM);
  k_gemm_bf16<<<gg, 256, 0, stream>>>(xbf, wvb, bv, vf, DIM, DIM);

  k_prep_q<<<(NBH * SEQ) / 8, 256, 0, stream>>>(qf, temp, qe, (unsigned int*)qbf);
  k_prep_kv<<<nX / 256, 256, 0, stream>>>(kf, vf, kbf, vT);

  dim3 ga(NBH, SEQ / 16);  // (32, 128)
  k_attn<<<ga, 32, 0, stream>>>(qbf, kbf, vT, hbuf);

  k_gate<<<NTOK / 8, 256, 0, stream>>>(hbuf, Wr, br, Ws, bs, hsb);

  k_gemm_bf16<<<gg, 256, 0, stream>>>(hsb, wpb, bp, out, DIM, DIM);
}
